// Prunus_85315230368055
// MI455X (gfx1250) — matrix operand order repaired and checked by reading, not compiled, not run on hardware
//
#include <hip/hip_runtime.h>
#include <hip/hip_bf16.h>

typedef __attribute__((ext_vector_type(16))) _Float16 v16h;
typedef __attribute__((ext_vector_type(8)))  float    v8f;

union Frag16 { v16h h; uint4 u[2]; };

#define TILE_M 128
#define TILE_N 64
#define TILE_K 32

// ---------------------------------------------------------------------------
// Async global->LDS copy (CDNA5 ASYNCcnt path). The low 32 bits of a generic
// pointer to __shared__ memory are the LDS byte offset (ISA 10.2 aperture).
// Async loads complete in order, so waiting ASYNCcnt <= N lets N newer copies
// stay in flight while guaranteeing older ones have landed (double buffering).
// ---------------------------------------------------------------------------
__device__ __forceinline__ void async_ld_b128(void* lds, const void* g)
{
    unsigned l = (unsigned)(size_t)lds;
    unsigned long long ga = (unsigned long long)(size_t)g;
    asm volatile("global_load_async_to_lds_b128 %0, %1, off"
                 :: "v"(l), "v"(ga) : "memory");
}
template<int N>
__device__ __forceinline__ void wait_async_le()
{
    asm volatile("s_wait_asynccnt %0" :: "i"(N) : "memory");
}

// ---------------------------------------------------------------------------
// WMMA GEMM:  C[M,N] (f32) = A[M,K] (f16) @ Wh[N,K]^T (f16) + bias
// Block = 256 threads (8 waves). Tile 128x64. Wave -> 32x32 (2x2 WMMA frags).
// Double-buffered LDS; tiles streamed with global_load_async_to_lds_b128,
// next tile in flight (3 asyncs/wave) while current tile feeds the WMMAs.
// ---------------------------------------------------------------------------
__global__ __launch_bounds__(256)
void k_gemm_wmma(const _Float16* __restrict__ A, const _Float16* __restrict__ Wh,
                 const float* __restrict__ bias, float* __restrict__ C,
                 int M, int N, int K)
{
    __shared__ _Float16 As[2][TILE_M][TILE_K];   // 16 KB
    __shared__ _Float16 Bs[2][TILE_N][TILE_K];   //  8 KB

    const int t    = threadIdx.x;
    const int lane = t & 31;
    const int wid  = t >> 5;
    const int wm   = (wid & 3) * 32;
    const int wn   = (wid >> 2) * 32;
    const int bm   = blockIdx.x * TILE_M;
    const int bn   = blockIdx.y * TILE_N;

    const v8f zero = {0.f,0.f,0.f,0.f,0.f,0.f,0.f,0.f};
    v8f acc[2][2];
    acc[0][0] = zero; acc[0][1] = zero; acc[1][0] = zero; acc[1][1] = zero;

    const int arow  = t >> 1;          // A: row t/2, 16 halves at (t&1)*16
    const int akoff = (t & 1) * 16;
    const int brow  = t >> 2;          // B: row t/4, 8 halves at (t&3)*8
    const int bkoff = (t & 3) * 8;

    const int kh = lane >> 4;
    const int lr = lane & 15;

    const _Float16* aln = A  + (size_t)(bm + arow) * K + akoff;
    const _Float16* bln = Wh + (size_t)(bn + brow) * K + bkoff;

    auto stage = [&](int buf, int k0) {
        async_ld_b128(&As[buf][arow][akoff],     aln + k0);
        async_ld_b128(&As[buf][arow][akoff + 8], aln + k0 + 8);
        async_ld_b128(&Bs[buf][brow][bkoff],     bln + k0);
    };

    const int nk = K / TILE_K;
    stage(0, 0);

    for (int i = 0; i < nk; ++i) {
        const int buf = i & 1;
        if (i + 1 < nk) {
            stage(buf ^ 1, (i + 1) * TILE_K);          // next tile in flight
            __builtin_prefetch(aln + (i + 2) * TILE_K, 0, 1);
            wait_async_le<3>();                        // current tile landed
        } else {
            wait_async_le<0>();
        }
        __syncthreads();

        // A 16x32: lane(l%16=row, l/16=kh): elems0-7 K=8kh.., elems8-15 K=16+8kh..
        Frag16 af[2];
        #pragma unroll
        for (int i2 = 0; i2 < 2; ++i2) {
            const _Float16* ap = &As[buf][wm + 16*i2 + lr][0];
            af[i2].u[0] = *(const uint4*)&ap[8*kh];
            af[i2].u[1] = *(const uint4*)&ap[16 + 8*kh];
        }
        // B 32x16: lane(l%16=col, l/16=kh): elems0-7 K=8kh.., elems8-15 K=16+8kh..
        Frag16 bf[2];
        #pragma unroll
        for (int j = 0; j < 2; ++j) {
            const _Float16* bp = &Bs[buf][wn + 16*j + lr][0];
            bf[j].u[0] = *(const uint4*)&bp[8*kh];
            bf[j].u[1] = *(const uint4*)&bp[16 + 8*kh];
        }

        #pragma unroll
        for (int i2 = 0; i2 < 2; ++i2)
            #pragma unroll
            for (int j = 0; j < 2; ++j)
                acc[i2][j] = __builtin_amdgcn_wmma_f32_16x16x32_f16(
                    false, af[i2].h, false, bf[j].h,
                    (short)0, acc[i2][j], false, false);

        __syncthreads();
    }

    #pragma unroll
    for (int i = 0; i < 2; ++i) {
        #pragma unroll
        for (int j = 0; j < 2; ++j) {
            const int col = bn + wn + 16*j + lr;
            const float bv = bias[col];
            #pragma unroll
            for (int v = 0; v < 8; ++v) {
                const int row = bm + wm + 16*i + 8*kh + v;
                C[(size_t)row * N + col] = acc[i][j][v] + bv;
            }
        }
    }
}

// ---------------------------------------------------------------------------
// Implicit-GEMM WMMA 3x3 pad-1 conv.  M = B*H*W pixels, N = Cout, K = Cin*9.
// A-tile: on-the-fly im2col gather into LDS; B-tile: double-buffered async
// f16 weight stream (1 async/wave in flight across the wait).
// in: f16 [B,Cin,H,W]; wh: f16 [Cout, Cin*9]; out raw f16 [B,Cout,H,W].
// ---------------------------------------------------------------------------
__global__ __launch_bounds__(256)
void k_conv_gemm_wmma(const _Float16* __restrict__ in, const _Float16* __restrict__ wh,
                      const float* __restrict__ bias, _Float16* __restrict__ out,
                      int Cin, int Cout, int H, int W)
{
    const int K  = Cin * 9;
    const int HW = H * W;
    __shared__ _Float16 As[TILE_M][TILE_K];
    __shared__ _Float16 Bs[2][TILE_N][TILE_K];

    const int t    = threadIdx.x;
    const int lane = t & 31;
    const int wid  = t >> 5;
    const int wm   = (wid & 3) * 32;
    const int wn   = (wid >> 2) * 32;
    const int bm   = blockIdx.x * TILE_M;   // pixel base
    const int bn   = blockIdx.y * TILE_N;   // out-channel base

    const v8f zero = {0.f,0.f,0.f,0.f,0.f,0.f,0.f,0.f};
    v8f acc[2][2];
    acc[0][0] = zero; acc[0][1] = zero; acc[1][0] = zero; acc[1][1] = zero;

    // A staging: thread t -> pixel row t/2, 16 K-elements at (t&1)*16
    const int arow  = t >> 1;
    const int akoff = (t & 1) * 16;
    const int apix  = bm + arow;
    const int an    = apix / HW;
    const int arem  = apix % HW;
    const int ay    = arem / W;
    const int ax    = arem % W;
    const _Float16* ibase = in + (size_t)an * Cin * HW;

    // B staging: thread t -> weight row t/4, 8 halves at (t&3)*8
    const int brow  = t >> 2;
    const int bkoff = (t & 3) * 8;
    const _Float16* bln = wh + (size_t)(bn + brow) * K + bkoff;

    const int kh = lane >> 4;
    const int lr = lane & 15;

    const int nk = K / TILE_K;
    async_ld_b128(&Bs[0][brow][bkoff], bln);

    for (int i = 0; i < nk; ++i) {
        const int buf = i & 1;
        const int k0  = i * TILE_K;

        // gather 3x3 patch elements (zero-padded) into the A tile
        #pragma unroll
        for (int hblk = 0; hblk < 2; ++hblk) {
            union { _Float16 h[8]; uint4 u; } pk;
            #pragma unroll
            for (int e = 0; e < 8; ++e) {
                const int kk = k0 + akoff + hblk * 8 + e;
                const int ci = kk / 9;
                const int kf = kk - ci * 9;
                const int iy = ay + kf / 3 - 1;
                const int ix = ax + kf % 3 - 1;
                float v = 0.f;
                if ((unsigned)iy < (unsigned)H && (unsigned)ix < (unsigned)W)
                    v = (float)ibase[(size_t)ci * HW + iy * W + ix];
                pk.h[e] = (_Float16)v;
            }
            *(uint4*)&As[arow][akoff + hblk * 8] = pk.u;
        }
        // stream next weight tile while this one is consumed
        if (i + 1 < nk) {
            async_ld_b128(&Bs[buf ^ 1][brow][bkoff], bln + (i + 1) * TILE_K);
            wait_async_le<1>();
        } else {
            wait_async_le<0>();
        }
        __syncthreads();

        Frag16 af[2];
        #pragma unroll
        for (int i2 = 0; i2 < 2; ++i2) {
            const _Float16* ap = &As[wm + 16*i2 + lr][0];
            af[i2].u[0] = *(const uint4*)&ap[8*kh];
            af[i2].u[1] = *(const uint4*)&ap[16 + 8*kh];
        }
        Frag16 bf[2];
        #pragma unroll
        for (int j = 0; j < 2; ++j) {
            const _Float16* bp = &Bs[buf][wn + 16*j + lr][0];
            bf[j].u[0] = *(const uint4*)&bp[8*kh];
            bf[j].u[1] = *(const uint4*)&bp[16 + 8*kh];
        }

        #pragma unroll
        for (int i2 = 0; i2 < 2; ++i2)
            #pragma unroll
            for (int j = 0; j < 2; ++j)
                acc[i2][j] = __builtin_amdgcn_wmma_f32_16x16x32_f16(
                    false, af[i2].h, false, bf[j].h,
                    (short)0, acc[i2][j], false, false);

        __syncthreads();
    }

    // scatter back to NCHW raw (pre-BN) f16
    #pragma unroll
    for (int i = 0; i < 2; ++i) {
        #pragma unroll
        for (int j = 0; j < 2; ++j) {
            const int co = bn + wn + 16*j + lr;
            const float bv = bias[co];
            #pragma unroll
            for (int v = 0; v < 8; ++v) {
                const int pix = bm + wm + 16*i + 8*kh + v;
                const int n   = pix / HW;
                const int rem = pix - n * HW;
                out[((size_t)n * Cout + co) * HW + rem] = (_Float16)(acc[i][j][v] + bv);
            }
        }
    }
}

// ---------------------------------------------------------------------------
// Direct 3x3 pad-1 conv for conv1 only (Cin=3, K=27), f16 in -> f16 raw out
// ---------------------------------------------------------------------------
__global__ __launch_bounds__(256)
void k_conv3x3(const _Float16* __restrict__ in, const float* __restrict__ w,
               const float* __restrict__ bias, _Float16* __restrict__ out,
               long long total, int Cin, int Cout, int H, int W)
{
    long long i = (long long)blockIdx.x * blockDim.x + threadIdx.x;
    if (i >= total) return;
    const int x  = (int)(i % W);
    long long r  = i / W;
    const int y  = (int)(r % H); r /= H;
    const int co = (int)(r % Cout);
    const long long n = r / Cout;

    float acc = bias[co];
    const float*    wc = w  + (long long)co * Cin * 9;
    const _Float16* ib = in + n * Cin * (long long)(H * W);
    for (int ci = 0; ci < Cin; ++ci) {
        const _Float16* ip = ib + (long long)ci * H * W;
        const float*    wp = wc + ci * 9;
        #pragma unroll
        for (int ky = 0; ky < 3; ++ky) {
            const int iy = y + ky - 1;
            if ((unsigned)iy >= (unsigned)H) continue;
            #pragma unroll
            for (int kx = 0; kx < 3; ++kx) {
                const int ix = x + kx - 1;
                if ((unsigned)ix >= (unsigned)W) continue;
                acc += (float)ip[iy * W + ix] * wp[ky * 3 + kx];
            }
        }
    }
    out[i] = (_Float16)acc;
}

// ---------------------------------------------------------------------------
// Per-channel sum/sumsq over NCHW f16 (one block per channel)
// ---------------------------------------------------------------------------
__global__ __launch_bounds__(256)
void k_stats_nchw(const _Float16* __restrict__ x, float* __restrict__ sum,
                  float* __restrict__ sumsq, int B, int C, int HW)
{
    const int c = blockIdx.x;
    float s = 0.f, q = 0.f;
    const long long tot = (long long)B * HW;
    for (long long i = threadIdx.x; i < tot; i += blockDim.x) {
        const long long b = i / HW;
        const int p = (int)(i % HW);
        const float v = (float)x[(b * C + c) * (long long)HW + p];
        s += v; q += v * v;
    }
    __shared__ float ss[256], qq[256];
    ss[threadIdx.x] = s; qq[threadIdx.x] = q;
    __syncthreads();
    for (int o = 128; o > 0; o >>= 1) {
        if ((int)threadIdx.x < o) { ss[threadIdx.x] += ss[threadIdx.x + o]; qq[threadIdx.x] += qq[threadIdx.x + o]; }
        __syncthreads();
    }
    if (threadIdx.x == 0) { sum[c] = ss[0]; sumsq[c] = qq[0]; }
}

// Per-column sum/sumsq over [B,C] f32 (one block per column)
__global__ __launch_bounds__(256)
void k_stats_col(const float* __restrict__ y, float* __restrict__ sum,
                 float* __restrict__ sumsq, int B, int C)
{
    const int c = blockIdx.x;
    float s = 0.f, q = 0.f;
    for (int b = threadIdx.x; b < B; b += blockDim.x) {
        const float v = y[(long long)b * C + c];
        s += v; q += v * v;
    }
    __shared__ float ss[256], qq[256];
    ss[threadIdx.x] = s; qq[threadIdx.x] = q;
    __syncthreads();
    for (int o = 128; o > 0; o >>= 1) {
        if ((int)threadIdx.x < o) { ss[threadIdx.x] += ss[threadIdx.x + o]; qq[threadIdx.x] += qq[threadIdx.x + o]; }
        __syncthreads();
    }
    if (threadIdx.x == 0) { sum[c] = ss[0]; sumsq[c] = qq[0]; }
}

__global__ void k_bn_finalize(const float* __restrict__ sum, const float* __restrict__ sumsq,
                              const float* __restrict__ g, const float* __restrict__ b,
                              float* __restrict__ scale, float* __restrict__ shift,
                              int C, float invCnt)
{
    const int c = blockIdx.x * blockDim.x + threadIdx.x;
    if (c >= C) return;
    const float m  = sum[c] * invCnt;
    const float v  = sumsq[c] * invCnt - m * m;
    const float is = rsqrtf(v + 1e-5f) * g[c];
    scale[c] = is;
    shift[c] = b[c] - m * is;
}

// BN + ReLU + 2x2 maxpool (NCHW f16 -> f16)
__global__ __launch_bounds__(256)
void k_bn_relu_pool(const _Float16* __restrict__ x, const float* __restrict__ scale,
                    const float* __restrict__ shift, _Float16* __restrict__ out,
                    long long total, int C, int H, int W)
{
    long long i = (long long)blockIdx.x * blockDim.x + threadIdx.x;
    if (i >= total) return;
    const int OW = W >> 1, OH = H >> 1;
    const int ox = (int)(i % OW);
    long long r  = i / OW;
    const int oy = (int)(r % OH); r /= OH;
    const int c  = (int)(r % C);
    const long long n = r / C;
    const _Float16* base = x + ((n * C + c) * (long long)H + 2 * oy) * W + 2 * ox;
    const float sc = scale[c], sh = shift[c];
    float m = 0.f;  // relu floor
    #pragma unroll
    for (int dy = 0; dy < 2; ++dy)
        #pragma unroll
        for (int dx = 0; dx < 2; ++dx) {
            const float v = (float)base[dy * W + dx] * sc + sh;
            m = fmaxf(m, v);
        }
    out[i] = (_Float16)m;
}

// BN + ReLU, NCHW f16 -> f16 (no pool)
__global__ __launch_bounds__(256)
void k_bn_relu_nchw(const _Float16* __restrict__ x, const float* __restrict__ scale,
                    const float* __restrict__ shift, _Float16* __restrict__ out,
                    long long total, int C, int HW)
{
    long long i = (long long)blockIdx.x * blockDim.x + threadIdx.x;
    if (i >= total) return;
    const int c = (int)((i / HW) % C);
    const float v = (float)x[i] * scale[c] + shift[c];
    out[i] = (_Float16)fmaxf(v, 0.f);
}

// BN + ReLU over [B,C] f32 -> f16
__global__ __launch_bounds__(256)
void k_bn_relu_2d(const float* __restrict__ y, const float* __restrict__ scale,
                  const float* __restrict__ shift, _Float16* __restrict__ out,
                  long long total, int C)
{
    long long i = (long long)blockIdx.x * blockDim.x + threadIdx.x;
    if (i >= total) return;
    const int c = (int)(i % C);
    const float v = y[i] * scale[c] + shift[c];
    out[i] = (_Float16)fmaxf(v, 0.f);
}

__global__ __launch_bounds__(256)
void k_relu_f32_to_f16(const float* __restrict__ y, _Float16* __restrict__ out, long long n)
{
    long long i = (long long)blockIdx.x * blockDim.x + threadIdx.x;
    if (i < n) out[i] = (_Float16)fmaxf(y[i], 0.f);
}

__global__ __launch_bounds__(256)
void k_f32_to_f16(const float* __restrict__ x, _Float16* __restrict__ out, long long n)
{
    long long i = (long long)blockIdx.x * blockDim.x + threadIdx.x;
    if (i < n) out[i] = (_Float16)x[i];
}

// Small FC: out[b,n] = bias[n] + sum_k in[b,k]*w[n,k]   (in f16, w/out f32)
__global__ __launch_bounds__(256)
void k_fc_small(const _Float16* __restrict__ in, const float* __restrict__ w,
                const float* __restrict__ bias, float* __restrict__ out,
                int B, int K, int N)
{
    const long long i = (long long)blockIdx.x * blockDim.x + threadIdx.x;
    if (i >= (long long)B * N) return;
    const int n = (int)(i % N);
    const long long b = i / N;
    const _Float16* ip = in + b * K;
    const float*    wp = w + (long long)n * K;
    float acc = bias[n];
    for (int k = 0; k < K; ++k) acc += (float)ip[k] * wp[k];
    out[i] = acc;
}

__device__ __forceinline__ float hash_u01(unsigned x)
{
    x ^= x >> 16; x *= 0x7feb352du; x ^= x >> 15; x *= 0x846ca68bu; x ^= x >> 16;
    return ((float)x + 1.0f) * 2.3283064e-10f;  // (0,1]
}

// Gumbel-softmax (tau=0.1) over P=2, categorical sample of partition
__global__ __launch_bounds__(256)
void k_gumbel_partition(const float* __restrict__ swl, float* __restrict__ idx_out,
                        float* __restrict__ ysum, int* __restrict__ pidx, int B)
{
    const int b = blockIdx.x * blockDim.x + threadIdx.x;
    if (b >= B) return;
    const float l0 = swl[2 * b], l1 = swl[2 * b + 1];
    const float g0 = -logf(-logf(hash_u01(3u * b + 1u)));
    const float g1 = -logf(-logf(hash_u01(3u * b + 2u)));
    const float a0 = (l0 + g0) * 10.0f, a1 = (l1 + g1) * 10.0f;  // /tau
    const float mx = fmaxf(a0, a1);
    const float e0 = expf(a0 - mx), e1 = expf(a1 - mx);
    const float inv = 1.0f / (e0 + e1);
    const float y0 = e0 * inv, y1 = e1 * inv;
    const float u  = hash_u01(3u * b + 3u) * (y0 + y1);
    const int p = (u <= y0) ? 0 : 1;
    pidx[b]    = p;
    ysum[b]    = y0 + y1;
    idx_out[b] = (float)p;
}

// Gather selected partition logits, scale by sum(y)
__global__ __launch_bounds__(256)
void k_select_class(const float* __restrict__ logitsP, const int* __restrict__ pidx,
                    const float* __restrict__ ysum, float* __restrict__ out, int B)
{
    const long long i = (long long)blockIdx.x * blockDim.x + threadIdx.x;
    if (i >= (long long)B * 10) return;
    const int c = (int)(i % 10);
    const int b = (int)(i / 10);
    const int p = pidx[b];
    out[i] = logitsP[((long long)p * B + b) * 10 + c] * ysum[b];
}

// ---------------------------------------------------------------------------
extern "C" void kernel_launch(void* const* d_in, const int* in_sizes, int n_in,
                              void* d_out, int out_size, void* d_ws, size_t ws_size,
                              hipStream_t stream)
{
    (void)in_sizes; (void)n_in; (void)out_size; (void)ws_size;
    constexpr int NB = 4096;

    const float* input   = (const float*)d_in[0];
    const float* conv1_w = (const float*)d_in[1];
    const float* conv1_b = (const float*)d_in[2];
    const float* bn1_g   = (const float*)d_in[3];
    const float* bn1_b   = (const float*)d_in[4];
    const float* conv2_w = (const float*)d_in[5];
    const float* conv2_b = (const float*)d_in[6];
    const float* bn2_g   = (const float*)d_in[7];
    const float* bn2_b   = (const float*)d_in[8];
    const float* conv3_w = (const float*)d_in[9];
    const float* conv3_b = (const float*)d_in[10];
    const float* bn3_g   = (const float*)d_in[11];
    const float* bn3_b   = (const float*)d_in[12];
    const float* pre_w   = (const float*)d_in[13];
    const float* pre_b   = (const float*)d_in[14];
    const float* bnp_g   = (const float*)d_in[15];
    const float* bnp_b   = (const float*)d_in[16];
    const float* disc_w  = (const float*)d_in[17];
    const float* disc_b  = (const float*)d_in[18];
    const float* bnd_g   = (const float*)d_in[19];
    const float* bnd_b   = (const float*)d_in[20];
    const float* dfc_w   = (const float*)d_in[21];
    const float* dfc_b   = (const float*)d_in[22];
    const float* sw_w    = (const float*)d_in[23];
    const float* sw_b    = (const float*)d_in[24];
    const float* p1_w    = (const float*)d_in[25];
    const float* p1_b    = (const float*)d_in[26];
    const float* p2_w    = (const float*)d_in[27];
    const float* p2_b    = (const float*)d_in[28];

    float* out      = (float*)d_out;
    float* dom_out  = out + (long long)NB * 10;
    float* pidx_out = out + (long long)NB * 12;

    // ---- workspace layout ----
    char* wsp = (char*)d_ws;
    auto alloc = [&](size_t bytes) -> void* {
        void* p = wsp; wsp += (bytes + 255) & ~(size_t)255; return p;
    };
    _Float16* x_h      = (_Float16*)alloc((size_t)NB * 3 * 1024 * 2);
    _Float16* c1       = (_Float16*)alloc((size_t)NB * 32 * 1024 * 2);
    _Float16* pool1    = (_Float16*)alloc((size_t)NB * 32 * 256 * 2);
    _Float16* c2       = (_Float16*)alloc((size_t)NB * 64 * 256 * 2);
    _Float16* pool2    = (_Float16*)alloc((size_t)NB * 64 * 64 * 2);
    _Float16* c3       = (_Float16*)alloc((size_t)NB * 128 * 64 * 2);
    _Float16* featc    = (_Float16*)alloc((size_t)NB * 8192 * 2);
    float*    pre_raw  = (float*)alloc((size_t)NB * 1024 * 4);
    _Float16* feat_h   = (_Float16*)alloc((size_t)NB * 1024 * 2);
    float*    disc_raw = (float*)alloc((size_t)NB * 384 * 4);
    _Float16* dpen_h   = (_Float16*)alloc((size_t)NB * 384 * 2);
    float*    h_raw    = (float*)alloc((size_t)2 * NB * 192 * 4);
    _Float16* h_h      = (_Float16*)alloc((size_t)2 * NB * 192 * 2);
    float*    logitsP  = (float*)alloc((size_t)2 * NB * 10 * 4);
    float*    swl      = (float*)alloc((size_t)NB * 2 * 4);
    float*    ysum     = (float*)alloc((size_t)NB * 4);
    int*      pidx     = (int*)alloc((size_t)NB * 4);
    float*    s_sum    = (float*)alloc(1024 * 4);
    float*    s_sq     = (float*)alloc(1024 * 4);
    float*    s_scale  = (float*)alloc(1024 * 4);
    float*    s_shift  = (float*)alloc(1024 * 4);
    // f16 weight copies for WMMA GEMMs
    _Float16* c2_wh    = (_Float16*)alloc((size_t)64 * 288 * 2);
    _Float16* c3_wh    = (_Float16*)alloc((size_t)128 * 576 * 2);
    _Float16* pre_wh   = (_Float16*)alloc((size_t)1024 * 8192 * 2);
    _Float16* disc_wh  = (_Float16*)alloc((size_t)384 * 1024 * 2);
    _Float16* p1_wh    = (_Float16*)alloc((size_t)2 * 192 * 1024 * 2);

    auto nblk = [](long long n) { return (unsigned)((n + 255) / 256); };

    // ---- weight conversions (f32 -> f16) ----
    k_f32_to_f16<<<nblk(64 * 288), 256, 0, stream>>>(conv2_w, c2_wh, 64 * 288);
    k_f32_to_f16<<<nblk(128 * 576), 256, 0, stream>>>(conv3_w, c3_wh, 128 * 576);
    k_f32_to_f16<<<nblk((long long)1024 * 8192), 256, 0, stream>>>(pre_w, pre_wh, (long long)1024 * 8192);
    k_f32_to_f16<<<nblk((long long)384 * 1024), 256, 0, stream>>>(disc_w, disc_wh, (long long)384 * 1024);
    k_f32_to_f16<<<nblk((long long)2 * 192 * 1024), 256, 0, stream>>>(p1_w, p1_wh, (long long)2 * 192 * 1024);

    // ---- stage 0: input f32 -> f16 ----
    const long long n_in0 = (long long)NB * 3 * 1024;
    k_f32_to_f16<<<nblk(n_in0), 256, 0, stream>>>(input, x_h, n_in0);

    // ---- conv1 (direct) + BN + ReLU + pool ----
    const long long t1 = (long long)NB * 32 * 1024;
    k_conv3x3<<<nblk(t1), 256, 0, stream>>>(x_h, conv1_w, conv1_b, c1, t1, 3, 32, 32, 32);
    k_stats_nchw<<<32, 256, 0, stream>>>(c1, s_sum, s_sq, NB, 32, 1024);
    k_bn_finalize<<<1, 32, 0, stream>>>(s_sum, s_sq, bn1_g, bn1_b, s_scale, s_shift, 32,
                                        1.0f / ((float)NB * 1024.f));
    const long long tp1 = (long long)NB * 32 * 256;
    k_bn_relu_pool<<<nblk(tp1), 256, 0, stream>>>(c1, s_scale, s_shift, pool1, tp1, 32, 32, 32);

    // ---- conv2 (implicit-GEMM WMMA) + BN + ReLU + pool ----
    {
        dim3 g((unsigned)((long long)NB * 256 / TILE_M), 64 / TILE_N);
        k_conv_gemm_wmma<<<g, 256, 0, stream>>>(pool1, c2_wh, conv2_b, c2, 32, 64, 16, 16);
    }
    k_stats_nchw<<<64, 256, 0, stream>>>(c2, s_sum, s_sq, NB, 64, 256);
    k_bn_finalize<<<1, 64, 0, stream>>>(s_sum, s_sq, bn2_g, bn2_b, s_scale, s_shift, 64,
                                        1.0f / ((float)NB * 256.f));
    const long long tp2 = (long long)NB * 64 * 64;
    k_bn_relu_pool<<<nblk(tp2), 256, 0, stream>>>(c2, s_scale, s_shift, pool2, tp2, 64, 16, 16);

    // ---- conv3 (implicit-GEMM WMMA) + BN + ReLU -> flat features [NB, 8192] ----
    {
        dim3 g((unsigned)((long long)NB * 64 / TILE_M), 128 / TILE_N);
        k_conv_gemm_wmma<<<g, 256, 0, stream>>>(pool2, c3_wh, conv3_b, c3, 64, 128, 8, 8);
    }
    const long long t3 = (long long)NB * 128 * 64;
    k_stats_nchw<<<128, 256, 0, stream>>>(c3, s_sum, s_sq, NB, 128, 64);
    k_bn_finalize<<<1, 128, 0, stream>>>(s_sum, s_sq, bn3_g, bn3_b, s_scale, s_shift, 128,
                                         1.0f / ((float)NB * 64.f));
    k_bn_relu_nchw<<<nblk(t3), 256, 0, stream>>>(c3, s_scale, s_shift, featc, t3, 128, 64);

    // ---- pre FC (WMMA): [4096,8192] x [1024,8192]^T ----
    {
        dim3 g(NB / TILE_M, 1024 / TILE_N);
        k_gemm_wmma<<<g, 256, 0, stream>>>(featc, pre_wh, pre_b, pre_raw, NB, 1024, 8192);
    }
    k_stats_col<<<1024, 256, 0, stream>>>(pre_raw, s_sum, s_sq, NB, 1024);
    k_bn_finalize<<<4, 256, 0, stream>>>(s_sum, s_sq, bnp_g, bnp_b, s_scale, s_shift, 1024,
                                         1.0f / (float)NB);
    const long long tf = (long long)NB * 1024;
    k_bn_relu_2d<<<nblk(tf), 256, 0, stream>>>(pre_raw, s_scale, s_shift, feat_h, tf, 1024);

    // ---- disc FC (WMMA): [4096,1024] x [384,1024]^T ----
    {
        dim3 g(NB / TILE_M, 384 / TILE_N);
        k_gemm_wmma<<<g, 256, 0, stream>>>(feat_h, disc_wh, disc_b, disc_raw, NB, 384, 1024);
    }
    k_stats_col<<<384, 256, 0, stream>>>(disc_raw, s_sum, s_sq, NB, 384);
    k_bn_finalize<<<2, 256, 0, stream>>>(s_sum, s_sq, bnd_g, bnd_b, s_scale, s_shift, 384,
                                         1.0f / (float)NB);
    const long long td = (long long)NB * 384;
    k_bn_relu_2d<<<nblk(td), 256, 0, stream>>>(disc_raw, s_scale, s_shift, dpen_h, td, 384);

    // ---- heads: domain_output and switch logits ----
    k_fc_small<<<nblk((long long)NB * 2), 256, 0, stream>>>(dpen_h, dfc_w, dfc_b, dom_out, NB, 384, 2);
    k_fc_small<<<nblk((long long)NB * 2), 256, 0, stream>>>(dpen_h, sw_w, sw_b, swl, NB, 384, 2);
    k_gumbel_partition<<<nblk(NB), 256, 0, stream>>>(swl, pidx_out, ysum, pidx, NB);

    // ---- expert MLPs (both partitions, dense): p1 via WMMA, p2 small ----
    for (int p = 0; p < 2; ++p) {
        dim3 g(NB / TILE_M, 192 / TILE_N);
        k_gemm_wmma<<<g, 256, 0, stream>>>(feat_h, p1_wh + (size_t)p * 192 * 1024,
                                           p1_b + p * 192, h_raw + (size_t)p * NB * 192,
                                           NB, 192, 1024);
    }
    const long long th = (long long)2 * NB * 192;
    k_relu_f32_to_f16<<<nblk(th), 256, 0, stream>>>(h_raw, h_h, th);
    for (int p = 0; p < 2; ++p) {
        k_fc_small<<<nblk((long long)NB * 10), 256, 0, stream>>>(
            h_h + (size_t)p * NB * 192, p2_w + (size_t)p * 10 * 192, p2_b + p * 10,
            logitsP + (size_t)p * NB * 10, NB, 192, 10);
    }
    k_select_class<<<nblk((long long)NB * 10), 256, 0, stream>>>(logitsP, pidx, ysum, out, NB);
}
